// Sonata_27874337751673
// MI455X (gfx1250) — matrix operand order repaired and checked by reading, not compiled, not run on hardware
//
#include <hip/hip_runtime.h>
#include <stdint.h>

// ---------------------------------------------------------------------------
// CDNA5 (gfx1250) implementation: bf16 WMMA GEMMs + fused patch attention,
// with async global->LDS tile staging (GLOBAL_LOAD_ASYNC_TO_LDS_B128).
//   N=131072 points, C=512, K=128 patch, H=8 heads, D=64.
// Pipeline:
//   0) convert feat fp32->bf16; convert+TRANSPOSE w_qkv / w_proj -> bf16 [n][k]
//   1) qkv GEMM (async double-buffered), gather by `order`      -> qkv_bf16
//   2) attention per (patch, head): QK^T, softmax, PV           -> attn_bf16
//   3) proj GEMM (async double-buffered), gather by `inverse`   -> out fp32
// ---------------------------------------------------------------------------

typedef __attribute__((ext_vector_type(16))) __bf16 v16bf;
typedef __attribute__((ext_vector_type(8)))  float  v8f;

#define WMMA_BF16(a, b, c) \
  __builtin_amdgcn_wmma_f32_16x16x32_bf16(false, (a), false, (b), (short)0, (c), false, false)

static constexpr int NPTS   = 131072;
static constexpr int C      = 512;
static constexpr int KPATCH = 128;
static constexpr int HEADS  = 8;
static constexpr int DHEAD  = 64;
static constexpr float SCALE = 0.125f;   // 64^-0.5

// ---- CDNA5 async global->LDS copy (ASYNCcnt-tracked, bypasses VGPRs) ------
__device__ __forceinline__ void async_copy_b128(const void* gptr, unsigned lds_off) {
  asm volatile("global_load_async_to_lds_b128 %0, %1, off"
               :: "v"(lds_off), "v"(gptr) : "memory");
}
__device__ __forceinline__ void wait_async0() {
  asm volatile("s_wait_asynccnt 0x0" ::: "memory");
}
__device__ __forceinline__ unsigned lds_off32(const void* p) {
  return (unsigned)(uintptr_t)p;   // flat LDS addr truncates to addr[31:0]
}

// ---------------------------------------------------------------------------
// fp32 -> bf16 conversion (elementwise)
// ---------------------------------------------------------------------------
__global__ __launch_bounds__(256) void cvt_f32_bf16(const float* __restrict__ in,
                                                    __bf16* __restrict__ out, int n) {
  int i = blockIdx.x * 256 + threadIdx.x;
  if (i < n) out[i] = (__bf16)in[i];
}

// fp32 [512][Nb] -> bf16 transposed [Nb][512] (tiny, one-time)
__global__ __launch_bounds__(256) void cvt_f32_bf16_t(const float* __restrict__ in,
                                                      __bf16* __restrict__ out,
                                                      int Nb, int n) {
  int i = blockIdx.x * 256 + threadIdx.x;
  if (i < n) {
    int k = i / Nb;
    int c = i - k * Nb;
    out[(size_t)c * 512 + k] = (__bf16)in[i];
  }
}

// ---------------------------------------------------------------------------
// Gathered bf16 GEMM: out[r, :] = A[gather[r], :] @ W + bias
//   A : [M, 512] bf16 row-major;  Wt : [Nb, 512] bf16 (W transposed)
//   Block tile 128x128, 8 waves (each 32x64 = 2x4 WMMA tiles), K-steps of 32.
//   Double-buffered LDS filled by async global->LDS b128 copies; one barrier
//   per k-step (s_wait_asynccnt + s_barrier).
// ---------------------------------------------------------------------------
template <bool OUT_F32>
__global__ __launch_bounds__(256)
void gemm_bf16_gather(const __bf16* __restrict__ A,
                      const int*    __restrict__ gather,
                      const __bf16* __restrict__ Wt,
                      const float*  __restrict__ bias,
                      void*         __restrict__ out,
                      int Nb) {
  __shared__ __bf16 As[2][128][40];   // [buf][m][k], row stride 80B (16B aligned)
  __shared__ __bf16 Bs[2][128][40];   // [buf][n][k]
  constexpr unsigned BUFSTRIDE = 128 * 40 * 2;   // 10240 bytes

  const int tid  = threadIdx.x;
  const int wave = tid >> 5;
  const int lane = tid & 31;
  const int wm   = wave >> 1;           // 0..3  (32-row slice)
  const int wn   = wave & 1;            // 0..1  (64-col slice)
  const int l16  = lane & 15;
  const bool lo  = (lane < 16);

  const int block_row = blockIdx.y * 128;
  const int block_col = blockIdx.x * 128;

  // ---- hoisted gather + per-thread streaming pointers ----
  const int ar   = tid >> 2;            // 0..63
  const int ac   = (tid & 3) * 8;       // 0,8,16,24
  const int src0 = gather[block_row + ar];
  const int src1 = gather[block_row + 64 + ar];
  const __bf16* a0 = A + (size_t)src0 * 512 + ac;
  const __bf16* a1 = A + (size_t)src1 * 512 + ac;
  const int bn   = tid >> 1;            // 0..127
  const int bk16 = (tid & 1) * 16;      // 0,16
  const __bf16* bp = Wt + (size_t)(block_col + bn) * 512 + bk16;

  const unsigned ldsA0 = lds_off32(&As[0][ar][ac]);
  const unsigned ldsA1 = lds_off32(&As[0][64 + ar][ac]);
  const unsigned ldsB0 = lds_off32(&Bs[0][bn][bk16]);
  const unsigned ldsB1 = ldsB0 + 16;

  v8f acc[2][4] = {};

  // ---- async-stage tile 0 ----
  async_copy_b128(a0, ldsA0);
  async_copy_b128(a1, ldsA1);
  async_copy_b128(bp, ldsB0);
  async_copy_b128(bp + 8, ldsB1);
  wait_async0();
  __syncthreads();

  for (int k0 = 0; k0 < 512; k0 += 32) {
    const int buf   = (k0 >> 5) & 1;
    const bool more = (k0 + 32) < 512;
    // ---- issue async copies for the next tile into the other buffer ----
    if (more) {
      const unsigned boff = (unsigned)(buf ^ 1) * BUFSTRIDE;
      async_copy_b128(a0 + k0 + 32, ldsA0 + boff);
      async_copy_b128(a1 + k0 + 32, ldsA1 + boff);
      async_copy_b128(bp + k0 + 32, ldsB0 + boff);
      async_copy_b128(bp + k0 + 40, ldsB1 + boff);
      if (k0 + 64 < 512) {               // one more tile ahead: L2 prefetch
        __builtin_prefetch(a0 + k0 + 64, 0, 1);
        __builtin_prefetch(bp + k0 + 64, 0, 1);
      }
    }

    // ---- A fragments: rows wm*32 + {0,16} ----
    v16bf afrag[2];
    const int arow = wm * 32 + l16;
    const int akb  = lo ? 0 : 16;
#pragma unroll
    for (int i = 0; i < 2; ++i) {
#pragma unroll
      for (int e = 0; e < 8; ++e) {
        afrag[i][e]     = As[buf][arow + i * 16][akb + e];
        afrag[i][e + 8] = As[buf][arow + i * 16][akb + 8 + e];
      }
    }
    // ---- B fragments + WMMA ----
#pragma unroll
    for (int j = 0; j < 4; ++j) {
      v16bf bfrag;
      const int bcol = wn * 64 + j * 16 + l16;
      const int bk   = lo ? 0 : 16;
#pragma unroll
      for (int e = 0; e < 16; ++e) bfrag[e] = Bs[buf][bcol][bk + e];
#pragma unroll
      for (int i = 0; i < 2; ++i) acc[i][j] = WMMA_BF16(afrag[i], bfrag, acc[i][j]);
    }
    wait_async0();     // next tile resident in LDS (this wave's copies)
    __syncthreads();   // ... and everyone else's
  }

  // ---- epilogue: bias add, store (fp32 or bf16) ----
  const int orow = block_row + wm * 32;
  const int ocol = block_col + wn * 64;
  const int rsel = lo ? 0 : 8;
#pragma unroll
  for (int i = 0; i < 2; ++i)
#pragma unroll
    for (int j = 0; j < 4; ++j)
#pragma unroll
      for (int v = 0; v < 8; ++v) {
        const int r = orow + i * 16 + v + rsel;
        const int c = ocol + j * 16 + l16;
        const float val = acc[i][j][v] + bias[c];
        if (OUT_F32)
          ((float*)out)[(size_t)r * Nb + c] = val;
        else
          ((__bf16*)out)[(size_t)r * Nb + c] = (__bf16)val;
      }
}

// ---------------------------------------------------------------------------
// Attention: one block per (patch, head). qkv serialized [N, 3*512] bf16 with
// row n holding [3][H][D]. Computes softmax(q k^T * scale) v -> out [N,512].
// 8 waves; wave w owns score rows 16w..16w+15 (full 16x128 slice in VGPRs).
// q/k staged via async global->LDS; v staged TRANSPOSED (vt[d][k]) so P@V
// B-fragments are contiguous b128s.
// ---------------------------------------------------------------------------
__global__ __launch_bounds__(256)
void attn_kernel(const __bf16* __restrict__ qkv, __bf16* __restrict__ out) {
  __shared__ __bf16 qk[2 * 128 * 64];   // q then k; reused as P[8][16][128]
  __shared__ __bf16 vt[64][136];        // v transposed [d][k], padded rows

  const int p    = blockIdx.x >> 3;
  const int h    = blockIdx.x & 7;
  const int tid  = threadIdx.x;
  const int wave = tid >> 5;
  const int lane = tid & 31;
  const int l16  = lane & 15;
  const bool lo  = (lane < 16);

  const size_t rowbase = (size_t)p * KPATCH * (3 * C);

  // ---- stage q, k via async copies; v transposed via VGPR path ----
#pragma unroll
  for (int m = 0; m < 2; ++m) {
#pragma unroll
    for (int ii = 0; ii < 4; ++ii) {
      const int idx = tid + 256 * ii;   // 0..1023
      const int r   = idx >> 3;
      const int q8  = (idx & 7) * 8;
      async_copy_b128(qkv + rowbase + (size_t)r * (3 * C) + m * C + h * DHEAD + q8,
                      lds_off32(qk + m * 128 * 64 + r * 64 + q8));
    }
  }
#pragma unroll
  for (int ii = 0; ii < 4; ++ii) {
    const int idx = tid + 256 * ii;
    const int r   = idx >> 3;           // patch row 0..127
    const int q8  = (idx & 7) * 8;      // d offset
    __bf16 tmp[8];
    *(uint4*)tmp = *(const uint4*)(qkv + rowbase + (size_t)r * (3 * C) +
                                   2 * C + h * DHEAD + q8);
#pragma unroll
    for (int i = 0; i < 8; ++i) vt[q8 + i][r] = tmp[i];
  }
  wait_async0();
  __syncthreads();

  // ---- S = q k^T for rows 16*wave .. 16*wave+15 ----
  const __bf16* qs = qk;
  const __bf16* ks = qk + 128 * 64;
  v8f accs[8] = {};
  v16bf qf[2];
  {
    const int arow = wave * 16 + l16;
    const int akb  = lo ? 0 : 16;
#pragma unroll
    for (int s = 0; s < 2; ++s)
#pragma unroll
      for (int e = 0; e < 8; ++e) {
        qf[s][e]     = qs[arow * 64 + s * 32 + akb + e];
        qf[s][e + 8] = qs[arow * 64 + s * 32 + akb + 8 + e];
      }
  }
#pragma unroll
  for (int t = 0; t < 8; ++t) {
#pragma unroll
    for (int s = 0; s < 2; ++s) {
      v16bf kf;
      const int kr = t * 16 + l16;
      const int kb = s * 32 + (lo ? 0 : 16);
#pragma unroll
      for (int e = 0; e < 16; ++e) kf[e] = ks[kr * 64 + kb + e];
      accs[t] = WMMA_BF16(qf[s], kf, accs[t]);
    }
  }
  __syncthreads();  // all waves done reading q/k before P overwrites them

  // ---- softmax (fp32) + write P (bf16) to recycled q/k LDS ----
  __bf16* ps = qk + wave * 16 * 128;   // this wave's [16][128]
  const int rsel = lo ? 0 : 8;
#pragma unroll
  for (int v = 0; v < 8; ++v) {
    float x[8];
    float m = -3.0e38f;
#pragma unroll
    for (int t = 0; t < 8; ++t) {
      x[t] = accs[t][v] * SCALE;
      m = fmaxf(m, x[t]);
    }
#pragma unroll
    for (int mask = 1; mask < 16; mask <<= 1) m = fmaxf(m, __shfl_xor(m, mask, 32));
    float sum = 0.0f;
#pragma unroll
    for (int t = 0; t < 8; ++t) {
      x[t] = __expf(x[t] - m);
      sum += x[t];
    }
#pragma unroll
    for (int mask = 1; mask < 16; mask <<= 1) sum += __shfl_xor(sum, mask, 32);
    const float rs = 1.0f / sum;
    const int r = rsel + v;
#pragma unroll
    for (int t = 0; t < 8; ++t) ps[r * 128 + t * 16 + l16] = (__bf16)(x[t] * rs);
  }
  __syncthreads();

  // ---- O = P @ V  (M=16, K=128, N=64); B frags contiguous from vt ----
  v8f acco[4] = {};
#pragma unroll
  for (int kb = 0; kb < 4; ++kb) {
    v16bf pf;
    const int prow = l16;
    const int pk   = kb * 32 + (lo ? 0 : 16);
#pragma unroll
    for (int e = 0; e < 8; ++e) {
      pf[e]     = ps[prow * 128 + pk + e];
      pf[e + 8] = ps[prow * 128 + pk + 8 + e];
    }
#pragma unroll
    for (int tn = 0; tn < 4; ++tn) {
      v16bf vf;
      const int d  = tn * 16 + l16;
      const int vk = kb * 32 + (lo ? 0 : 16);
#pragma unroll
      for (int e = 0; e < 16; ++e) vf[e] = vt[d][vk + e];
      acco[tn] = WMMA_BF16(pf, vf, acco[tn]);
    }
  }

  // ---- store serialized attention output [N, 512] bf16 ----
  const size_t obase = (size_t)p * KPATCH * C;
#pragma unroll
  for (int tn = 0; tn < 4; ++tn)
#pragma unroll
    for (int v = 0; v < 8; ++v) {
      const int r = wave * 16 + v + rsel;
      const int c = h * DHEAD + tn * 16 + l16;
      out[obase + (size_t)r * C + c] = (__bf16)acco[tn][v];
    }
}

// ---------------------------------------------------------------------------
// Host-side launcher
// ---------------------------------------------------------------------------
extern "C" void kernel_launch(void* const* d_in, const int* in_sizes, int n_in,
                              void* d_out, int out_size, void* d_ws, size_t ws_size,
                              hipStream_t stream) {
  const float* feat   = (const float*)d_in[0];
  const float* w_qkv  = (const float*)d_in[1];
  const float* b_qkv  = (const float*)d_in[2];
  const float* w_proj = (const float*)d_in[3];
  const float* b_proj = (const float*)d_in[4];
  const int*   order  = (const int*)d_in[5];
  const int*   inv    = (const int*)d_in[6];
  float*       out    = (float*)d_out;

  char* ws = (char*)d_ws;
  __bf16* feat_bf   = (__bf16*)ws;  ws += (size_t)NPTS * C * 2;
  __bf16* wqkv_bft  = (__bf16*)ws;  ws += (size_t)C * 3 * C * 2;   // [1536][512]
  __bf16* wproj_bft = (__bf16*)ws;  ws += (size_t)C * C * 2;       // [512][512]
  __bf16* qkv_bf    = (__bf16*)ws;  ws += (size_t)NPTS * 3 * C * 2;
  __bf16* attn_bf   = (__bf16*)ws;  ws += (size_t)NPTS * C * 2;
  (void)ws_size; (void)n_in; (void)in_sizes; (void)out_size;

  // 0) fp32 -> bf16 conversions (weights transposed to [n][k])
  {
    int n0 = NPTS * C;          // 67,108,864
    int n1 = C * 3 * C;         // 786,432
    int n2 = C * C;             // 262,144
    cvt_f32_bf16<<<(n0 + 255) / 256, 256, 0, stream>>>(feat, feat_bf, n0);
    cvt_f32_bf16_t<<<(n1 + 255) / 256, 256, 0, stream>>>(w_qkv, wqkv_bft, 3 * C, n1);
    cvt_f32_bf16_t<<<(n2 + 255) / 256, 256, 0, stream>>>(w_proj, wproj_bft, C, n2);
  }

  // 1) qkv = feat[order] @ w_qkv + b_qkv   (M=131072, K=512, N=1536) -> bf16
  gemm_bf16_gather<false><<<dim3(1536 / 128, NPTS / 128), 256, 0, stream>>>(
      feat_bf, order, wqkv_bft, b_qkv, qkv_bf, 3 * C);

  // 2) patch attention  (1024 patches x 8 heads) -> attn_bf16 [N, 512]
  attn_kernel<<<(NPTS / KPATCH) * HEADS, 256, 0, stream>>>(qkv_bf, attn_bf);

  // 3) out = attn[inverse] @ w_proj + b_proj  (M=131072, K=512, N=512) -> f32
  gemm_bf16_gather<true><<<dim3(512 / 128, NPTS / 128), 256, 0, stream>>>(
      attn_bf, inv, wproj_bft, b_proj, out, C);
}
